// CAWN2_12421045420080
// MI455X (gfx1250) — compile-verified, matrix operand order checked by reading, not run on hardware
//
#include <hip/hip_runtime.h>
#include <hip/hip_bf16.h>

// ---------------- problem constants (from reference) ----------------
#define BDIM 4096           // B
#define KN   32             // K neighbors
#define FD   64             // F
#define HD   128            // H
#define DINC 128            // DIN = 2F
#define NH3  384            // 3H
#define G3B  (3*BDIM)       // 12288 row-groups

// ---------------- WMMA helper types ----------------
typedef __attribute__((ext_vector_type(16))) _Float16 v16h;
typedef __attribute__((ext_vector_type(8)))  _Float16 v8h;
typedef __attribute__((ext_vector_type(8)))  float    v8f;

__device__ __forceinline__ v8f wmma_f16(v16h a, v16h b, v8f c) {
  // D = A(16x32 f16) x B(32x16 f16) + C(16x16 f32)
  return __builtin_amdgcn_wmma_f32_16x16x32_f16(
      /*neg_a=*/false, a, /*neg_b=*/false, b,
      /*c_mod=*/(short)0, c, /*reuse_a=*/false, /*reuse_b=*/false);
}

// Load 16-bit A fragment from a row-major LDS row.
// Lane layout (ISA 7.12.2, 16-bit A 16x32): lanes 0-15 hold K {0..7,16..23},
// lanes 16-31 hold K {8..15,24..31} of row M = lane%16.
// koff = ktile*32 + (lane>>4)*8 (in halves); two 16-byte LDS reads.
__device__ __forceinline__ v16h lda_row(const _Float16* row, int koff) {
  union { v16h v; v8h h[2]; } u;
  u.h[0] = *(const v8h*)(row + koff);
  u.h[1] = *(const v8h*)(row + koff + 16);
  return u.v;
}

// ---------------- kernel 0: pack f32 weights into f16 B-fragments ----------------
// W is [Nrows][Kdim] row-major; GEMM is out = X @ W^T, so B[k][n] = W[n][k].
// Packed layout: [(nt*ktiles + kt)][lane 0..31][16 halves], lane L holds
// column n = nt*16 + L%16 with K pattern mirroring the A-operand striping.
__global__ __launch_bounds__(256)
void pack_b_f16(const float* __restrict__ W, _Float16* __restrict__ out,
                int Nrows, int Kdim) {
  int tid = blockIdx.x * blockDim.x + threadIdx.x;
  int ktiles = Kdim >> 5;
  int total = (Nrows >> 4) * ktiles * 512;
  if (tid >= total) return;
  int j    = tid & 15;
  int lane = (tid >> 4) & 31;
  int tile = tid >> 9;
  int kt   = tile % ktiles;
  int nt   = tile / ktiles;
  int n     = nt * 16 + (lane & 15);
  int base0 = (lane >> 4) * 8;
  int k     = kt * 32 + ((j < 8) ? (base0 + j) : (16 + base0 + (j - 8)));
  out[tid] = (_Float16)W[(long)n * Kdim + k];
}

// ---------------- kernel 1: fused time-enc + gather + GRU + masked mean ----------------
// One block per group g in [0, 3B). 256 threads = 8 waves.
__global__ __launch_bounds__(256)
void gru_agg_kernel(const float* __restrict__ cut_time,
                    const int*   __restrict__ ngh_id,
                    const int*   __restrict__ e_idx,
                    const float* __restrict__ ngh_ts,
                    const float* __restrict__ hidden,
                    const float* __restrict__ e_feat,
                    const float* __restrict__ basis_freq,
                    const float* __restrict__ phase,
                    const _Float16* __restrict__ packWih,
                    const _Float16* __restrict__ packWhh,
                    const float* __restrict__ b_ih,
                    const float* __restrict__ b_hh,
                    float* __restrict__ agg_out) {
  __shared__ _Float16 xs[KN * DINC];   // 32x128 f16 x-matrix        (8 KB)
  __shared__ _Float16 hs[KN * HD];     // 32x128 f16 h-matrix        (8 KB)
  __shared__ float    gx[KN * NH3];    // 32x384 f32 x@Wih^T         (48 KB)
  __shared__ float    gh[KN * NH3];    // 32x384 f32 h@Whh^T         (48 KB)
  __shared__ float    maskv[KN];
  __shared__ float    part[2][HD];

  const int g = blockIdx.x;
  const int t = threadIdx.x;
  const float ct = cut_time[g & (BDIM - 1)];   // ct = tile(cut_time,3)[g]

  // ---- phase 1: build x (e_feat gather | cos time embedding) and h in f16 ----
  {
    const int j  = t >> 3;           // neighbor row 0..31
    const int cs = (t & 7) * 16;     // 16-column segment
    const int gi = g * KN + j;
    if (cs < FD) {
      const float* er = e_feat + (long)e_idx[gi] * FD + cs;
      #pragma unroll
      for (int i = 0; i < 16; ++i) xs[j * DINC + cs + i] = (_Float16)er[i];
    } else {
      const float dt = ct - ngh_ts[gi];
      const int c0 = cs - FD;
      #pragma unroll
      for (int i = 0; i < 16; ++i)
        xs[j * DINC + cs + i] = (_Float16)__cosf(dt * basis_freq[c0 + i] + phase[c0 + i]);
    }
    const float* hr = hidden + (long)gi * HD + cs;
    #pragma unroll
    for (int i = 0; i < 16; ++i) hs[j * HD + cs + i] = (_Float16)hr[i];
    if (t < KN) maskv[t] = (ngh_id[g * KN + t] != 0) ? 1.0f : 0.0f;
  }
  __syncthreads();

  // ---- phase 2: gx = x @ Wih^T, gh = h @ Whh^T via WMMA ----
  // A operands (x and h tiles) are hoisted into registers ONCE per wave
  // (16 fragments = 128 VGPRs) and reused across all three N-tiles; only the
  // B weight fragments stream from L2 (hot: packed weights total < 200 KB).
  {
    const int wave = t >> 5, lane = t & 31;
    const int mrow  = lane & 15;
    const int base0 = (lane >> 4) * 8;
    const v16h* pWih = (const v16h*)packWih;   // 32B per lane-fragment
    const v16h* pWhh = (const v16h*)packWhh;

    v16h Ax0[4], Ax1[4], Ah0[4], Ah1[4];
    #pragma unroll
    for (int kt = 0; kt < 4; ++kt) {
      const int koff = kt * 32 + base0;
      Ax0[kt] = lda_row(&xs[mrow * DINC],        koff);
      Ax1[kt] = lda_row(&xs[(16 + mrow) * DINC], koff);
      Ah0[kt] = lda_row(&hs[mrow * HD],          koff);
      Ah1[kt] = lda_row(&hs[(16 + mrow) * HD],   koff);
    }

    for (int nti = 0; nti < 3; ++nti) {
      const int nt = wave * 3 + nti;           // N-tile 0..23
      v8f ax0 = {}, ax1 = {}, ah0 = {}, ah1 = {};
      #pragma unroll
      for (int kt = 0; kt < 4; ++kt) {
        v16h bih = pWih[(nt * 4 + kt) * 32 + lane];
        v16h bhh = pWhh[(nt * 4 + kt) * 32 + lane];
        ax0 = wmma_f16(Ax0[kt], bih, ax0);
        ax1 = wmma_f16(Ax1[kt], bih, ax1);
        ah0 = wmma_f16(Ah0[kt], bhh, ah0);
        ah1 = wmma_f16(Ah1[kt], bhh, ah1);
      }
      // C layout: VGPR i -> M = (lane>>4)*8 + i, N = lane%16
      const int n0 = nt * 16 + (lane & 15);
      const int mb = (lane >> 4) * 8;
      #pragma unroll
      for (int i = 0; i < 8; ++i) {
        gx[(mb + i) * NH3 + n0]        = ax0[i];
        gx[(16 + mb + i) * NH3 + n0]   = ax1[i];
        gh[(mb + i) * NH3 + n0]        = ah0[i];
        gh[(16 + mb + i) * NH3 + n0]   = ah1[i];
      }
    }
  }
  __syncthreads();

  // ---- phase 3: GRU nonlinearity + masked mean over K ----
  {
    const int c    = t & (HD - 1);
    const int half = t >> 7;
    float cnt = 0.f;
    #pragma unroll
    for (int j = 0; j < KN; ++j) cnt += maskv[j];
    const float denom = fmaxf(cnt, 1.0f);
    const float bir = b_ih[c],          bhr = b_hh[c];
    const float biz = b_ih[HD + c],     bhz = b_hh[HD + c];
    const float bin = b_ih[2 * HD + c], bhn = b_hh[2 * HD + c];
    float sum = 0.f;
    #pragma unroll
    for (int jj = 0; jj < 16; ++jj) {
      const int j = half * 16 + jj;
      const float rr = gx[j * NH3 + c] + bir + gh[j * NH3 + c] + bhr;
      const float zz = gx[j * NH3 + HD + c] + biz + gh[j * NH3 + HD + c] + bhz;
      const float xn = gx[j * NH3 + 2 * HD + c] + bin;
      const float hn = gh[j * NH3 + 2 * HD + c] + bhn;
      const float r  = 1.f / (1.f + __expf(-rr));
      const float z  = 1.f / (1.f + __expf(-zz));
      const float nn = tanhf(xn + r * hn);
      const float hp = (float)hs[j * HD + c];
      sum += maskv[j] * ((1.f - z) * nn + z * hp);
    }
    part[half][c] = sum;
    __syncthreads();
    if (t < HD) agg_out[(long)g * HD + c] = (part[0][c] + part[1][c]) / denom;
  }
}

// ---------------- kernel 2: emb = relu([node_raw | agg] @ W_out^T + b_out) ----------------
// One block handles 32 rows; M=32 (2 tiles) x N=64 (4 tiles), K=192 (6 ktiles).
__global__ __launch_bounds__(256)
void emb_kernel(const int* __restrict__ src_ids,
                const int* __restrict__ tgt_ids,
                const int* __restrict__ bad_ids,
                const float* __restrict__ n_feat,
                const float* __restrict__ agg,
                const _Float16* __restrict__ packWout,
                const float* __restrict__ b_out,
                float* __restrict__ emb) {
  __shared__ _Float16 in_s[32 * 192];   // 12 KB
  const int blk = blockIdx.x;
  const int t   = threadIdx.x;
  {
    const int j  = t >> 3;
    const int cs = (t & 7) * 24;
    const int row = blk * 32 + j;
    int id;
    if (row < BDIM)            id = src_ids[row];
    else if (row < 2 * BDIM)   id = tgt_ids[row - BDIM];
    else                       id = bad_ids[row - 2 * BDIM];
    const float* nf = n_feat + (long)id * FD;
    const float* ag = agg + (long)row * HD;
    #pragma unroll
    for (int i = 0; i < 24; ++i) {
      const int cc = cs + i;
      const float v = (cc < FD) ? nf[cc] : ag[cc - FD];
      in_s[j * 192 + cc] = (_Float16)v;
    }
  }
  __syncthreads();
  const int wave = t >> 5, lane = t & 31;
  const int mt = wave & 1, ntile = wave >> 1;   // 8 waves -> 8 output tiles
  const int mrow  = lane & 15;
  const int base0 = (lane >> 4) * 8;
  const v16h* pW = (const v16h*)packWout;
  v8f acc = {};
  #pragma unroll
  for (int kt = 0; kt < 6; ++kt) {
    v16h a = lda_row(&in_s[(mt * 16 + mrow) * 192], kt * 32 + base0);
    v16h b = pW[(ntile * 6 + kt) * 32 + lane];
    acc = wmma_f16(a, b, acc);
  }
  const int col = ntile * 16 + (lane & 15);
  const float bo = b_out[col];
  const int mb = (lane >> 4) * 8;
  #pragma unroll
  for (int i = 0; i < 8; ++i) {
    const int row = blk * 32 + mt * 16 + mb + i;
    emb[(long)row * FD + col] = fmaxf(acc[i] + bo, 0.f);
  }
}

// ---------------- kernel 3: merge MLP (tiny, VALU) ----------------
__global__ __launch_bounds__(256)
void merge_kernel(const float* __restrict__ emb,
                  const float* __restrict__ fc1_w,
                  const float* __restrict__ fc1_b,
                  const float* __restrict__ fc2_w,
                  const float* __restrict__ fc2_b,
                  float* __restrict__ out) {
  const int tid = blockIdx.x * blockDim.x + threadIdx.x;
  if (tid >= 2 * BDIM) return;
  const int r = tid >> 1, sel = tid & 1;       // sel 0: tgt (pos), 1: bad (neg)
  const float* x1 = emb + (long)r * FD;
  const float* x2 = emb + (long)(BDIM * (1 + sel) + r) * FD;
  float o = fc2_b[0];
  for (int f = 0; f < FD; ++f) {
    float s = fc1_b[f];
    const float* w = fc1_w + f * 2 * FD;
    #pragma unroll 8
    for (int i = 0; i < FD; ++i) s += w[i] * x1[i];
    #pragma unroll 8
    for (int i = 0; i < FD; ++i) s += w[FD + i] * x2[i];
    o += fc2_w[f] * fmaxf(s, 0.f);
  }
  out[r * 2 + sel] = o;
}

// ---------------- host launcher ----------------
extern "C" void kernel_launch(void* const* d_in, const int* in_sizes, int n_in,
                              void* d_out, int out_size, void* d_ws, size_t ws_size,
                              hipStream_t stream) {
  const int*   src_ids    = (const int*)  d_in[0];
  const int*   tgt_ids    = (const int*)  d_in[1];
  const int*   bad_ids    = (const int*)  d_in[2];
  const float* cut_time   = (const float*)d_in[3];
  const int*   ngh_id     = (const int*)  d_in[4];
  const int*   e_idx      = (const int*)  d_in[5];
  const float* ngh_ts     = (const float*)d_in[6];
  const float* hidden     = (const float*)d_in[7];
  const float* n_feat     = (const float*)d_in[8];
  const float* e_feat     = (const float*)d_in[9];
  const float* basis_freq = (const float*)d_in[10];
  const float* phase      = (const float*)d_in[11];
  const float* W_ih       = (const float*)d_in[12];
  const float* W_hh       = (const float*)d_in[13];
  const float* b_ih       = (const float*)d_in[14];
  const float* b_hh       = (const float*)d_in[15];
  const float* W_out      = (const float*)d_in[16];
  const float* b_out      = (const float*)d_in[17];
  const float* fc1_w      = (const float*)d_in[18];
  const float* fc1_b      = (const float*)d_in[19];
  const float* fc2_w      = (const float*)d_in[20];
  const float* fc2_b      = (const float*)d_in[21];
  float* out = (float*)d_out;

  // workspace layout (bytes), all offsets 256-aligned
  char* ws = (char*)d_ws;
  _Float16* packWih  = (_Float16*)(ws);                       //  98304 B
  _Float16* packWhh  = (_Float16*)(ws + 98304);               //  98304 B
  _Float16* packWout = (_Float16*)(ws + 196608);              //  24576 B
  float*    agg      = (float*)   (ws + 221184);              // 3B*128*4 = 6291456 B
  float*    emb      = (float*)   (ws + 221184 + 6291456);    // 3B*64*4  = 3145728 B

  // 0) pack weights into f16 WMMA B-fragments (tiny; stays hot in L2)
  pack_b_f16<<<(49152 + 255) / 256, 256, 0, stream>>>(W_ih, packWih, NH3, DINC);
  pack_b_f16<<<(49152 + 255) / 256, 256, 0, stream>>>(W_hh, packWhh, NH3, HD);
  pack_b_f16<<<(12288 + 255) / 256, 256, 0, stream>>>(W_out, packWout, FD, FD + HD);

  // 1) fused gather + time-encode + GRU + masked mean  (dominant: WMMA)
  gru_agg_kernel<<<G3B, 256, 0, stream>>>(cut_time, ngh_id, e_idx, ngh_ts, hidden,
                                          e_feat, basis_freq, phase,
                                          packWih, packWhh, b_ih, b_hh, agg);

  // 2) embedding projection (WMMA)
  emb_kernel<<<G3B / 32, 256, 0, stream>>>(src_ids, tgt_ids, bad_ids, n_feat, agg,
                                           packWout, b_out, emb);

  // 3) merge MLP -> (B, 2)
  merge_kernel<<<(2 * BDIM + 255) / 256, 256, 0, stream>>>(emb, fc1_w, fc1_b,
                                                           fc2_w, fc2_b, out);
}
